// VRWKV_Classification_31086973288814
// MI455X (gfx1250) — matrix operand order repaired and checked by reading, not compiled, not run on hardware
//
#include <hip/hip_runtime.h>
#include <hip/hip_bf16.h>

// ---------------------------------------------------------------------------
// VRWKV block for MI455X (gfx1250, wave32).
//  * GEMMs: v_wmma_f32_16x16x32_f16, fp32 accum, 128x128x32 block tile,
//    8 waves x (2x4) 16x16 tiles, double-buffered LDS.
//  * Staging: Tensor Data Mover (tensor_load_to_lds) with LDS padding,
//    pure-f16 operands (producers emit f16; weights pre-transposed to f16).
//  * Elementwise fully fused; bi-WKV parallel over B*C channels.
// ---------------------------------------------------------------------------

typedef __attribute__((ext_vector_type(16))) _Float16 v16h;
typedef __attribute__((ext_vector_type(8)))  float    v8f;
typedef unsigned int u32x4 __attribute__((ext_vector_type(4)));
typedef int          i32x4 __attribute__((ext_vector_type(4)));
typedef int          i32x8 __attribute__((ext_vector_type(8)));

namespace vrwkv {

constexpr int kB   = 128;
constexpr int kT   = 196;
constexpr int kC   = 768;
constexpr int kHID = 3072;
constexpr int kH   = 14;
constexpr int kW   = 14;
constexpr int kM   = kB * kT;        // 25088 rows
constexpr float kEPS = 1e-5f;

#define DEVFN static __device__ __forceinline__

// ------------------------------- reductions --------------------------------

DEVFN float block_sum(float v, float* sbuf) {
    const int tid = threadIdx.x;
#pragma unroll
    for (int o = 16; o > 0; o >>= 1) v += __shfl_down(v, o, 32);
    if ((tid & 31) == 0) sbuf[tid >> 5] = v;
    __syncthreads();
    if (tid == 0) {
        float s = 0.f;
#pragma unroll
        for (int i = 0; i < 8; ++i) s += sbuf[i];
        sbuf[0] = s;
    }
    __syncthreads();
    float r = sbuf[0];
    __syncthreads();
    return r;
}

// ------------------------------ LayerNorm ----------------------------------
// One row per 256-thread block. nC in {768, 3072} -> 3 or 12 elems/thread.
// SCALE: multiply output by `scale` row. OT: float (residual path) or
// _Float16 (GEMM-A operands). Supports in == out via register caching.

template <bool SCALE, typename OT>
__global__ __launch_bounds__(256) void ln_one_kernel(
    const float* __restrict__ in, OT* __restrict__ out,
    const float* __restrict__ g, const float* __restrict__ b,
    const float* __restrict__ scale, int nC)
{
    __shared__ float sbuf[8];
    const int row = blockIdx.x;
    const int tid = threadIdx.x;
    const int ept = nC >> 8;
    const float* ir = in + (size_t)row * nC;

    float vals[12];
    float s = 0.f;
    for (int i = 0; i < ept; ++i) { vals[i] = ir[i * 256 + tid]; s += vals[i]; }
    const float mean = block_sum(s, sbuf) / (float)nC;
    float vs = 0.f;
    for (int i = 0; i < ept; ++i) { float d = vals[i] - mean; vs += d * d; }
    const float rstd = rsqrtf(block_sum(vs, sbuf) / (float)nC + kEPS);

    OT* orow = out + (size_t)row * nC;
    for (int i = 0; i < ept; ++i) {
        const int c = i * 256 + tid;
        float y = (vals[i] - mean) * rstd * g[c] + b[c];
        if (SCALE) y *= scale[(size_t)row * nC + c];
        orow[c] = (OT)y;
    }
}

// Fused LN0 -> LN1 (C=768). x0 fp32 (residual stream), xi fp32 (mix input).
__global__ __launch_bounds__(256) void ln0_ln1_kernel(
    const float* __restrict__ x, float* __restrict__ x0, float* __restrict__ xi,
    const float* __restrict__ g0, const float* __restrict__ b0,
    const float* __restrict__ g1, const float* __restrict__ b1)
{
    __shared__ float sbuf[8];
    const int row = blockIdx.x;
    const int tid = threadIdx.x;
    const float* ir = x + (size_t)row * kC;

    float vals[3];
    float s = 0.f;
#pragma unroll
    for (int i = 0; i < 3; ++i) { vals[i] = ir[i * 256 + tid]; s += vals[i]; }
    float mean = block_sum(s, sbuf) * (1.0f / kC);
    float vs = 0.f;
#pragma unroll
    for (int i = 0; i < 3; ++i) { float d = vals[i] - mean; vs += d * d; }
    float rstd = rsqrtf(block_sum(vs, sbuf) * (1.0f / kC) + kEPS);
#pragma unroll
    for (int i = 0; i < 3; ++i) {
        const int c = i * 256 + tid;
        vals[i] = (vals[i] - mean) * rstd * g0[c] + b0[c];
        x0[(size_t)row * kC + c] = vals[i];
    }
    s = 0.f;
#pragma unroll
    for (int i = 0; i < 3; ++i) s += vals[i];
    mean = block_sum(s, sbuf) * (1.0f / kC);
    vs = 0.f;
#pragma unroll
    for (int i = 0; i < 3; ++i) { float d = vals[i] - mean; vs += d * d; }
    rstd = rsqrtf(block_sum(vs, sbuf) * (1.0f / kC) + kEPS);
#pragma unroll
    for (int i = 0; i < 3; ++i) {
        const int c = i * 256 + tid;
        xi[(size_t)row * kC + c] = (vals[i] - mean) * rstd * g1[c] + b1[c];
    }
}

// --------------------------- q_shift + token-mix ---------------------------
// Shift computed inline; outputs are f16 (they only feed GEMM A operands).

template <int NMIX>
__global__ __launch_bounds__(256) void shift_mix_kernel(
    const float* __restrict__ xi,
    _Float16* __restrict__ ok, _Float16* __restrict__ ov,
    _Float16* __restrict__ orr,
    const float* __restrict__ mk, const float* __restrict__ mv,
    const float* __restrict__ mr)
{
    const int row = blockIdx.x;                 // flattened (b, t)
    const int bI = row / kT;
    const int t  = row - bI * kT;
    const int h  = t / kW;
    const int w  = t - h * kW;
    const float* xrow = xi + (size_t)row * kC;

    for (int c = threadIdx.x; c < kC; c += 256) {
        const int grp = c / (kC / 4);
        int hh = h, ww = w;
        bool valid = true;
        if      (grp == 0) { ww = w - 1; valid = (ww >= 0); }
        else if (grp == 1) { ww = w + 1; valid = (ww < kW); }
        else if (grp == 2) { hh = h - 1; valid = (hh >= 0); }
        else               { hh = h + 1; valid = (hh < kH); }
        const float xxv = valid
            ? xi[((size_t)bI * kT + hh * kW + ww) * kC + c] : 0.f;
        const float xiv = xrow[c];
        const size_t off = (size_t)row * kC + c;
        { const float a = mk[c]; ok[off] = (_Float16)(xiv * a + xxv * (1.f - a)); }
        if (NMIX == 3) { const float a = mv[c]; ov[off] = (_Float16)(xiv * a + xxv * (1.f - a)); }
        { const float a = mr[c]; orr[off] = (_Float16)(xiv * a + xxv * (1.f - a)); }
    }
}

// ----------------- weight convert + transpose (fp32 -> f16) ----------------
// W: (K, N) fp32 row-major  ->  Wt: (N, K) f16 row-major. 64x64 LDS tiles,
// both global sides coalesced, 32B vector writes.

__global__ __launch_bounds__(256) void transpose_cvt_kernel(
    const float* __restrict__ W, _Float16* __restrict__ Wt, int K, int N)
{
    __shared__ float tile[64][65];
    const int k0 = blockIdx.x * 64;
    const int n0 = blockIdx.y * 64;
#pragma unroll
    for (int i = 0; i < 4; ++i) {
        const int idx = threadIdx.x + i * 256;   // 1024 float4 slots
        const int r  = idx >> 4;                 // 16 float4 per row
        const int cq = idx & 15;
        const float4 f = *(const float4*)(W + (size_t)(k0 + r) * N + n0 + cq * 4);
        tile[r][cq * 4 + 0] = f.x; tile[r][cq * 4 + 1] = f.y;
        tile[r][cq * 4 + 2] = f.z; tile[r][cq * 4 + 3] = f.w;
    }
    __syncthreads();
    const int n  = threadIdx.x >> 2;             // 64 rows of Wt
    const int ch = (threadIdx.x & 3) * 16;       // 16 k's per thread
    _Float16 o[16];
#pragma unroll
    for (int j = 0; j < 16; ++j) o[j] = (_Float16)tile[ch + j][n];
    float4* dst = (float4*)(Wt + (size_t)(n0 + n) * K + k0 + ch);
    dst[0] = ((const float4*)o)[0];
    dst[1] = ((const float4*)o)[1];
}

// ------------------------------ WMMA GEMM ----------------------------------
// D = epilogue(A @ Bt^T).  A: (M,K) f16, Bt: (N,K) f16 (pre-transposed).
// Block tile 128x128, K-step 32. Staging via Tensor Data Mover into padded
// double-buffered LDS; waves 0/1 own the A/B descriptors.

constexpr int BM = 128, BN = 128, BK = 32;
constexpr int LDA = BK + 8;   // f16 stride 40 -> 80B rows (TDM pad: 4DW/16DW)
constexpr int LDB = BK + 8;

DEVFN v8f wmma_16x16x32(v16h a, v16h b, v8f c) {
    return __builtin_amdgcn_wmma_f32_16x16x32_f16(
        false, a, false, b, (short)0, c, false, false);
}

// A fragment (16x32, §7.12.2): lane l -> row l%16; l/16 selects K-halves.
DEVFN v16h load_frag_A(const _Float16* lds, int mbase) {
    const int lane = threadIdx.x & 31;
    const _Float16* p = lds + (size_t)(mbase + (lane & 15)) * LDA;
    const int kh = (lane >> 4) * 8;
    v16h r;
#pragma unroll
    for (int i = 0; i < 8; ++i) r[i] = p[kh + i];
#pragma unroll
    for (int i = 0; i < 8; ++i) r[8 + i] = p[16 + kh + i];
    return r;
}

// B fragment (32x16): lane l -> col l%16; l/16 selects K-halves (same as A).
DEVFN v16h load_frag_B(const _Float16* lds, int nbase) {
    const int lane = threadIdx.x & 31;
    const _Float16* p = lds + (size_t)(nbase + (lane & 15)) * LDB + (lane >> 4) * 8;
    v16h r;
#pragma unroll
    for (int i = 0; i < 8; ++i) r[i] = p[i];
#pragma unroll
    for (int i = 0; i < 8; ++i) r[8 + i] = p[16 + i];
    return r;
}

// Issue one 2-D TDM descriptor: tile 32 (contig f16) x 128 rows from a
// row-major (rows x row_len) f16 matrix into LDS with 16B pad per 64B row
// (-> LDS row stride 80B == LDA f16). D# packing per CDNA5 ISA.
DEVFN void tdm_issue_2d(const _Float16* gptr, unsigned lds_byte,
                        int row_len, int rows)
{
    const unsigned long long ga = (unsigned long long)(uintptr_t)gptr;
    u32x4 g0;
    g0[0] = 1u;                                      // count=1, user mode
    g0[1] = lds_byte;                                // lds_addr
    g0[2] = (unsigned)(ga & 0xffffffffu);            // global_addr lo
    g0[3] = (unsigned)((ga >> 32) & 0x1ffffffu)      // global_addr hi (57b)
          | (2u << 30);                              // type=2 ("image")
    i32x8 g1;
    g1[0] = (int)((1u << 16)      // data_size = 2 bytes
                | (1u << 20)      // pad_enable
                | (3u << 22)      // pad_interval: every 16 DWORDs (64B)
                | (3u << 25));    // pad_amount: 4 DWORDs (16B)
    g1[1] = (int)(((unsigned)row_len & 0xffffu) << 16);            // dim0 lo
    g1[2] = (int)((((unsigned)row_len >> 16) & 0xffffu)
                | (((unsigned)rows & 0xffffu) << 16));             // dim0 hi | dim1 lo
    g1[3] = (int)(((((unsigned)rows >> 16) & 0xffffu))
                | ((unsigned)BK << 16));                           // dim1 hi | tile_dim0
    g1[4] = 128;                                                   // tile_dim1 (tile_dim2=0)
    g1[5] = (int)(unsigned)row_len;                                // dim0_stride lo
    g1[6] = 0;                                                     // stride hi | dim1_stride lo
    g1[7] = 0;
    const i32x4 gz = {0, 0, 0, 0};
#if defined(__clang_major__) && __clang_major__ >= 23
    const i32x8 gz8 = {0, 0, 0, 0, 0, 0, 0, 0};
    __builtin_amdgcn_tensor_load_to_lds(g0, g1, gz, gz, gz8, 0);
#else
    __builtin_amdgcn_tensor_load_to_lds(g0, g1, gz, gz, 0);
#endif
}

// Epilogues: 0 plain, 1 sigmoid, 2 relu^2, 3 aux1+acc, 4 aux2 + aux1*acc
template <int EPI>
__global__ __launch_bounds__(256) void gemm_kernel(
    const _Float16* __restrict__ A, const _Float16* __restrict__ Bt,
    float* __restrict__ D,
    const float* __restrict__ aux1, const float* __restrict__ aux2,
    int M, int N, int K)
{
    __shared__ _Float16 As[2][BM * LDA];
    __shared__ _Float16 Bs[2][BN * LDB];

    const int tid  = threadIdx.x;
    const int wave = tid >> 5;
    const int wm   = wave & 3;        // 4 m-slots of 32
    const int wn   = wave >> 2;       // 2 n-slots of 64
    const int block_m = blockIdx.y * BM;
    const int block_n = blockIdx.x * BN;

    v8f acc[2][4] = {};
    const int nk = K / BK;

    if (wave == 0)
        tdm_issue_2d(A + (size_t)block_m * K, (unsigned)(uintptr_t)&As[0][0], K, M);
    if (wave == 1)
        tdm_issue_2d(Bt + (size_t)block_n * K, (unsigned)(uintptr_t)&Bs[0][0], K, N);
    __builtin_amdgcn_s_wait_tensorcnt(0);
    __syncthreads();

    for (int kt = 0; kt < nk; ++kt) {
        const int cur = kt & 1;
        if (kt + 1 < nk) {
            const int kb = (kt + 1) * BK;
            if (wave == 0)
                tdm_issue_2d(A + (size_t)block_m * K + kb,
                             (unsigned)(uintptr_t)&As[cur ^ 1][0], K, M);
            if (wave == 1)
                tdm_issue_2d(Bt + (size_t)block_n * K + kb,
                             (unsigned)(uintptr_t)&Bs[cur ^ 1][0], K, N);
        }
        const v16h a0 = load_frag_A(As[cur], wm * 32 + 0);
        const v16h a1 = load_frag_A(As[cur], wm * 32 + 16);
        v16h b[4];
#pragma unroll
        for (int j = 0; j < 4; ++j) b[j] = load_frag_B(Bs[cur], wn * 64 + j * 16);
#pragma unroll
        for (int j = 0; j < 4; ++j) {
            acc[0][j] = wmma_16x16x32(a0, b[j], acc[0][j]);
            acc[1][j] = wmma_16x16x32(a1, b[j], acc[1][j]);
        }
        __builtin_amdgcn_s_wait_tensorcnt(0);
        __syncthreads();
    }

    // C/D layout (§7.12.2): lane l -> N = l%16; VGPR r -> M = r + 8*(l/16).
    const int lane = tid & 31;
#pragma unroll
    for (int i = 0; i < 2; ++i) {
#pragma unroll
        for (int j = 0; j < 4; ++j) {
            const int n  = block_n + wn * 64 + j * 16 + (lane & 15);
            const int m0 = block_m + wm * 32 + i * 16 + ((lane >> 4) << 3);
#pragma unroll
            for (int r = 0; r < 8; ++r) {
                const size_t off = (size_t)(m0 + r) * N + n;
                const float a = acc[i][j][r];
                float o;
                if constexpr (EPI == 0)      o = a;
                else if constexpr (EPI == 1) o = 1.f / (1.f + __expf(-a));
                else if constexpr (EPI == 2) { const float t = a > 0.f ? a : 0.f; o = t * t; }
                else if constexpr (EPI == 3) o = aux1[off] + a;
                else                         o = aux2[off] + aux1[off] * a;
                D[off] = o;
            }
        }
    }
}

// ---------------------------- bidirectional WKV ----------------------------

__global__ __launch_bounds__(256) void bi_wkv_kernel(
    const float* __restrict__ k, const float* __restrict__ v,
    const float* __restrict__ decay, const float* __restrict__ first,
    float* __restrict__ fa, float* __restrict__ fb, float* __restrict__ out)
{
    const int idx = blockIdx.x * 256 + threadIdx.x;   // < B*C
    const int bI = idx / kC;
    const int c  = idx - bI * kC;
    const size_t base = (size_t)bI * kT * kC + c;

    const float w  = decay[c] / (float)kT;
    const float u  = first[c] / (float)kT;
    const float d  = __expf(-w);
    const float ew = __expf(w);

    float kmax = -3.4e38f;
    for (int t = 0; t < kT; ++t)
        kmax = fmaxf(kmax, k[base + (size_t)t * kC]);

    float a = 0.f, b = 0.f;
    for (int t = 0; t < kT; ++t) {
        const size_t o = base + (size_t)t * kC;
        fa[o] = a; fb[o] = b;                          // carry BEFORE update
        const float ek = __expf(k[o] - kmax);
        a = d * (a + ek * v[o]);
        b = d * (b + ek);
    }
    a = 0.f; b = 0.f;
    for (int t = kT - 1; t >= 0; --t) {
        const size_t o = base + (size_t)t * kC;
        const float kk = k[o], vv = v[o];
        const float ek = __expf(kk - kmax);
        const float eu = __expf(u + kk - kmax);
        const float num = ew * (fa[o] + a) + eu * vv;
        const float den = ew * (fb[o] + b) + eu;
        out[o] = num / den;
        a = d * (a + ek * vv);
        b = d * (b + ek);
    }
}

} // namespace vrwkv

// ------------------------------- launcher ----------------------------------

extern "C" void kernel_launch(void* const* d_in, const int* in_sizes, int n_in,
                              void* d_out, int out_size, void* d_ws, size_t ws_size,
                              hipStream_t stream) {
    using namespace vrwkv;
    (void)in_sizes; (void)n_in; (void)out_size; (void)ws_size;

    const float* x         = (const float*)d_in[0];
    const float* ln0_g     = (const float*)d_in[1];
    const float* ln0_b     = (const float*)d_in[2];
    const float* ln1_g     = (const float*)d_in[3];
    const float* ln1_b     = (const float*)d_in[4];
    const float* ln2_g     = (const float*)d_in[5];
    const float* ln2_b     = (const float*)d_in[6];
    const float* att_mix_k = (const float*)d_in[7];
    const float* att_mix_v = (const float*)d_in[8];
    const float* att_mix_r = (const float*)d_in[9];
    const float* att_decay = (const float*)d_in[10];
    const float* att_first = (const float*)d_in[11];
    const float* att_Wk    = (const float*)d_in[12];
    const float* att_Wv    = (const float*)d_in[13];
    const float* att_Wr    = (const float*)d_in[14];
    const float* att_Wo    = (const float*)d_in[15];
    const float* att_kn_g  = (const float*)d_in[16];
    const float* att_kn_b  = (const float*)d_in[17];
    const float* ffn_mix_k = (const float*)d_in[18];
    const float* ffn_mix_r = (const float*)d_in[19];
    const float* ffn_Wk    = (const float*)d_in[20];
    const float* ffn_Wv    = (const float*)d_in[21];
    const float* ffn_Wr    = (const float*)d_in[22];
    const float* ffn_kn_g  = (const float*)d_in[23];
    const float* ffn_kn_b  = (const float*)d_in[24];
    float* out = (float*)d_out;

    // ---- workspace bump allocator (256B aligned) ----
    char* wsp = (char*)d_ws;
    auto allocF = [&](size_t n) {
        float* r = (float*)wsp;
        wsp += (n * sizeof(float) + 255) & ~(size_t)255;
        return r;
    };
    auto allocH = [&](size_t n) {
        _Float16* r = (_Float16*)wsp;
        wsp += (n * sizeof(_Float16) + 255) & ~(size_t)255;
        return r;
    };
    const size_t MC = (size_t)kM * kC;
    const size_t MH = (size_t)kM * kHID;
    // fp32 stream + scan buffers
    float* s0   = allocF(MC);   // x0, then x1 (residual)
    float* s1   = allocF(MC);   // xi, then xi2
    float* kf   = allocF(MC);   // k
    float* vf   = allocF(MC);   // v
    float* srf  = allocF(MC);   // sr, then sigmoid(xr2 @ ffn_Wr)
    float* faf  = allocF(MC);   // forward prefix a
    float* fbf  = allocF(MC);   // forward prefix b
    float* rwk  = allocF(MC);   // rwkv
    float* hidf = allocF(MH);   // relu^2 pre-LN
    // f16 GEMM-A operands
    _Float16* hxk = allocH(MC); // xk, then xk2
    _Float16* hxv = allocH(MC); // xv
    _Float16* hxr = allocH(MC); // xr, then xr2
    _Float16* hs8 = allocH(MC); // sr * LN(rwkv)
    _Float16* hhd = allocH(MH); // LN(relu^2) f16
    // f16 pre-transposed weights (N, K)
    _Float16* wkT = allocH((size_t)kC * kC);
    _Float16* wvT = allocH((size_t)kC * kC);
    _Float16* wrT = allocH((size_t)kC * kC);
    _Float16* woT = allocH((size_t)kC * kC);
    _Float16* fkT = allocH((size_t)kHID * kC);  // from (C,HID)
    _Float16* fvT = allocH((size_t)kC * kHID);  // from (HID,C)
    _Float16* frT = allocH((size_t)kC * kC);

    const dim3 blk(256);
    const dim3 gC(kC / BN, kM / BM);       // (6, 196)
    const dim3 gH(kHID / BN, kM / BM);     // (24, 196)
    const dim3 tCC(kC / 64, kC / 64);      // weight transposes
    const dim3 tCH(kC / 64, kHID / 64);
    const dim3 tHC(kHID / 64, kC / 64);

    // 0) weights -> f16, transposed (once per call; ~16MB, negligible)
    transpose_cvt_kernel<<<tCC, blk, 0, stream>>>(att_Wk, wkT, kC, kC);
    transpose_cvt_kernel<<<tCC, blk, 0, stream>>>(att_Wv, wvT, kC, kC);
    transpose_cvt_kernel<<<tCC, blk, 0, stream>>>(att_Wr, wrT, kC, kC);
    transpose_cvt_kernel<<<tCC, blk, 0, stream>>>(att_Wo, woT, kC, kC);
    transpose_cvt_kernel<<<tCH, blk, 0, stream>>>(ffn_Wk, fkT, kC, kHID);
    transpose_cvt_kernel<<<tHC, blk, 0, stream>>>(ffn_Wv, fvT, kHID, kC);
    transpose_cvt_kernel<<<tCC, blk, 0, stream>>>(ffn_Wr, frT, kC, kC);

    // 1) x0 = LN0(x); xi = LN1(x0)
    ln0_ln1_kernel<<<kM, blk, 0, stream>>>(x, s0, s1, ln0_g, ln0_b, ln1_g, ln1_b);
    // 2) q_shift + att mixes (f16 out)
    shift_mix_kernel<3><<<kM, blk, 0, stream>>>(s1, hxk, hxv, hxr,
                                                att_mix_k, att_mix_v, att_mix_r);
    // 3-5) k, v, sr
    gemm_kernel<0><<<gC, blk, 0, stream>>>(hxk, wkT, kf, nullptr, nullptr, kM, kC, kC);
    gemm_kernel<0><<<gC, blk, 0, stream>>>(hxv, wvT, vf, nullptr, nullptr, kM, kC, kC);
    gemm_kernel<1><<<gC, blk, 0, stream>>>(hxr, wrT, srf, nullptr, nullptr, kM, kC, kC);
    // 6) rwkv
    bi_wkv_kernel<<<(kB * kC) / 256, blk, 0, stream>>>(kf, vf, att_decay, att_first,
                                                       faf, fbf, rwk);
    // 7) hs8 = sr * LN(rwkv)   (f16)
    ln_one_kernel<true, _Float16><<<kM, blk, 0, stream>>>(rwk, hs8, att_kn_g, att_kn_b,
                                                          srf, kC);
    // 8) x1 = x0 + hs8 @ Wo    (in place on s0)
    gemm_kernel<3><<<gC, blk, 0, stream>>>(hs8, woT, s0, s0, nullptr, kM, kC, kC);
    // 9) xi2 = LN2(x1)
    ln_one_kernel<false, float><<<kM, blk, 0, stream>>>(s0, s1, ln2_g, ln2_b, nullptr, kC);
    // 10) ffn mixes (f16 out; reuse hxk/hxr)
    shift_mix_kernel<2><<<kM, blk, 0, stream>>>(s1, hxk, nullptr, hxr,
                                                ffn_mix_k, nullptr, ffn_mix_r);
    // 11) hidf = relu(xk2 @ ffn_Wk)^2
    gemm_kernel<2><<<gH, blk, 0, stream>>>(hxk, fkT, hidf, nullptr, nullptr, kM, kHID, kC);
    // 12) hhd = LN(hidf) (f16)
    ln_one_kernel<false, _Float16><<<kM, blk, 0, stream>>>(hidf, hhd, ffn_kn_g, ffn_kn_b,
                                                           nullptr, kHID);
    // 13) srf = sigmoid(xr2 @ ffn_Wr)
    gemm_kernel<1><<<gC, blk, 0, stream>>>(hxr, frT, srf, nullptr, nullptr, kM, kC, kC);
    // 14) out = x1 + srf * (hhd @ ffn_Wv)
    gemm_kernel<4><<<gC, blk, 0, stream>>>(hhd, fvT, out, srf, s0, kM, kC, kHID);
}
